// SelfAttention_20873541058659
// MI455X (gfx1250) — matrix operand order repaired and checked by reading, not compiled, not run on hardware
//
#include <hip/hip_runtime.h>
#include <hip/hip_bf16.h>

// ---------------------------------------------------------------------------
// Self-attention + FFN block, algebraically collapsed:
//   Vsum = x @ Wvsum            (Wvsum[e,d] = sum_c Wv[e, c*64+d])
//   A    = reshape(Vsum, 8x128x1024)   (flat reinterpretation, free)
//   Y    = A @ Wo + bo
//   H    = relu(Y @ W1 + b1)
//   F    = H @ W2 + b2
//   out[n, q*128 + r, :] = F[n*128 + r, :]   for q in 0..15
// All GEMMs via v_wmma_f32_16x16x32_bf16 (bf16 in, fp32 accumulate),
// register-blocked 32x64 per wave (8 wmma per 12 b128 loads per K-step).
// ---------------------------------------------------------------------------

#define EMBED   1024
#define HEADS   16
#define HEAD_DIM 64
#define NBATCH  8
#define SEQ     2048
#define ROWS    (NBATCH * SEQ)     // 16384
#define FF      4096
#define AROWS   (NBATCH * 128)     // 1024 distinct activation rows

typedef float  v8f     __attribute__((ext_vector_type(8)));
typedef __bf16 bf16x16 __attribute__((ext_vector_type(16)));

// ---------------- conversion / folding kernels ----------------

__global__ void cvt_f32_bf16_x4(const float* __restrict__ in,
                                __bf16* __restrict__ out, int n4) {
  int i = blockIdx.x * blockDim.x + threadIdx.x;
  if (i >= n4) return;
  float4 f = reinterpret_cast<const float4*>(in)[i];
  __bf16* o = out + 4 * (size_t)i;
  o[0] = (__bf16)f.x; o[1] = (__bf16)f.y; o[2] = (__bf16)f.z; o[3] = (__bf16)f.w;
}

// Wt[n][k] = (bf16) W[k][n];  W is K x N row-major, Wt is N x K row-major.
__global__ void transpose_f32_bf16(const float* __restrict__ W,
                                   __bf16* __restrict__ Wt, int K, int N) {
  int idx = blockIdx.x * blockDim.x + threadIdx.x;
  if (idx >= K * N) return;
  int n = idx / K;
  int k = idx - n * K;
  Wt[(size_t)n * K + k] = (__bf16)W[(size_t)k * N + n];
}

// Bt[d][e] = sum_c Wv[e][c*64 + d]  -> pre-transposed (64 x 1024) bf16.
__global__ void wvsum_t_bf16(const float* __restrict__ Wv,
                             __bf16* __restrict__ Bt) {
  int idx = blockIdx.x * blockDim.x + threadIdx.x;
  if (idx >= HEAD_DIM * EMBED) return;
  int d = idx >> 10;             // 0..63
  int e = idx & (EMBED - 1);     // 0..1023
  float s = 0.0f;
#pragma unroll
  for (int c = 0; c < HEADS; ++c) s += Wv[(size_t)e * EMBED + c * HEAD_DIM + d];
  Bt[idx] = (__bf16)s;
}

// ---------------- WMMA bf16 GEMM, 32x64 tile per wave ----------------
// C(MxN) = A(MxK) @ Bt(NxK)^T + bias[n]
// One wave computes a 32x64 tile: 2 M-subtiles x 4 N-subtiles of 16x16.
// Per K-step (32): 4 A-fragment loads + 8 B-fragment loads -> 8 wmma.
// OMODE 0: bf16 store (row-major MxN). OMODE 1: fp32 store broadcast 16x into
//          the (8,2048,1024) output: row gm -> batch gm>>7, r=gm&127,
//          written to rows q*128+r for q=0..15.
template<bool RELU, int OMODE>
__global__ __launch_bounds__(32)
void gemm_wmma_bf16(const __bf16* __restrict__ A,
                    const __bf16* __restrict__ Bt,
                    const float* __restrict__ bias,
                    void* __restrict__ Cout,
                    int M, int N, int K) {
  const int lane = threadIdx.x;          // 0..31, full wave, EXEC all-ones
  const int half = lane >> 4;            // 0 or 1
  const int lid  = lane & 15;
  const int tm   = blockIdx.y << 5;      // 32-row tile
  const int tn   = blockIdx.x << 6;      // 64-col tile

  // A (16-bit, 16x32): lane holds row M=lid; elems 0..7 = K[half*8 .. +7],
  //                    elems 8..15 = K[half*8+16 .. +23]
  const __bf16* aptr0 = A  + (size_t)(tm + lid) * K + (half << 3);
  const __bf16* aptr1 = aptr0 + (size_t)16 * K;
  // B (16-bit, 32x16): lane holds col N=lid; elems 0..7 = K[half*8 .. +7],
  //                    elems 8..15 = K[half*8+16 .. +23]
  const __bf16* bptr0 = Bt + (size_t)(tn + lid) * K + (half << 3);

  v8f acc[2][4] = {};
  union Pack { uint4 q[2]; bf16x16 v; };

#pragma unroll 2
  for (int k0 = 0; k0 < K; k0 += 32) {
    Pack a[2], b[4];
    a[0].q[0] = *reinterpret_cast<const uint4*>(aptr0 + k0);
    a[0].q[1] = *reinterpret_cast<const uint4*>(aptr0 + k0 + 16);
    a[1].q[0] = *reinterpret_cast<const uint4*>(aptr1 + k0);
    a[1].q[1] = *reinterpret_cast<const uint4*>(aptr1 + k0 + 16);
#pragma unroll
    for (int j = 0; j < 4; ++j) {
      const __bf16* bp = bptr0 + (size_t)j * 16 * K + k0;
      b[j].q[0] = *reinterpret_cast<const uint4*>(bp);
      b[j].q[1] = *reinterpret_cast<const uint4*>(bp + 16);
    }
#pragma unroll
    for (int i = 0; i < 2; ++i)
#pragma unroll
      for (int j = 0; j < 4; ++j)
        acc[i][j] = __builtin_amdgcn_wmma_f32_16x16x32_bf16(
            /*neg_a=*/false, a[i].v, /*neg_b=*/false, b[j].v,
            /*c_mod=*/(short)0, acc[i][j], /*reuse_a=*/false, /*reuse_b=*/false);
  }

#pragma unroll
  for (int j = 0; j < 4; ++j) {
    const int   n  = tn + (j << 4) + lid;
    const float bv = bias ? bias[n] : 0.0f;
#pragma unroll
    for (int i = 0; i < 2; ++i) {
      if (OMODE == 0) {
        __bf16* C = (__bf16*)Cout;
#pragma unroll
        for (int jj = 0; jj < 8; ++jj) {
          int m = tm + (i << 4) + (half << 3) + jj;  // VGPR jj -> M = jj + half*8
          float v = acc[i][j][jj] + bv;
          if (RELU) v = fmaxf(v, 0.0f);
          C[(size_t)m * N + n] = (__bf16)v;
        }
      } else {
        float* C = (float*)Cout;                     // (8, 2048, 1024) fp32
#pragma unroll
        for (int jj = 0; jj < 8; ++jj) {
          int gm = tm + (i << 4) + (half << 3) + jj; // 0..1023
          float v = acc[i][j][jj] + bv;
          if (RELU) v = fmaxf(v, 0.0f);
          float* orow = C + (((size_t)(gm >> 7) * SEQ + (gm & 127)) << 10) + n;
#pragma unroll
          for (int q = 0; q < 16; ++q) orow[(size_t)q * (128 * EMBED)] = v;
        }
      }
    }
  }
}

// ---------------------------------------------------------------------------

extern "C" void kernel_launch(void* const* d_in, const int* in_sizes, int n_in,
                              void* d_out, int out_size, void* d_ws, size_t ws_size,
                              hipStream_t stream) {
  const float* x  = (const float*)d_in[0];
  // d_in[1] = Wq, d_in[2] = Wk : provably unused (softmax rows sum to 1).
  const float* Wv = (const float*)d_in[3];
  const float* Wo = (const float*)d_in[4];
  const float* bo = (const float*)d_in[5];
  const float* W1 = (const float*)d_in[6];
  const float* b1 = (const float*)d_in[7];
  const float* W2 = (const float*)d_in[8];
  const float* b2 = (const float*)d_in[9];
  float* out = (float*)d_out;

  char* ws = (char*)d_ws;
  size_t off = 0;
  auto carve = [&](size_t bytes) -> char* {
    char* p = ws + off;
    off += (bytes + 255) & ~(size_t)255;
    return p;
  };
  __bf16* xb   = (__bf16*)carve((size_t)ROWS * EMBED * 2);     // 32 MiB
  __bf16* WvT  = (__bf16*)carve((size_t)HEAD_DIM * EMBED * 2); // 128 KiB (64x1024)
  __bf16* WoT  = (__bf16*)carve((size_t)EMBED * EMBED * 2);    // 2 MiB (1024x1024)
  __bf16* W1T  = (__bf16*)carve((size_t)FF * EMBED * 2);       // 8 MiB (4096x1024)
  __bf16* W2T  = (__bf16*)carve((size_t)EMBED * FF * 2);       // 8 MiB (1024x4096)
  __bf16* Aact = (__bf16*)carve((size_t)AROWS * EMBED * 2);    // 2 MiB (1024x1024)
  __bf16* Yb   = (__bf16*)carve((size_t)AROWS * EMBED * 2);    // 2 MiB
  __bf16* Hb   = (__bf16*)carve((size_t)AROWS * FF * 2);       // 8 MiB

  const int BT = 256;

  // --- precision conversion / weight folding ---
  {
    int n4 = ROWS * EMBED / 4;
    cvt_f32_bf16_x4<<<(n4 + BT - 1) / BT, BT, 0, stream>>>(x, xb, n4);
  }
  transpose_f32_bf16<<<(EMBED * EMBED + BT - 1) / BT, BT, 0, stream>>>(Wo, WoT, EMBED, EMBED);
  transpose_f32_bf16<<<(EMBED * FF    + BT - 1) / BT, BT, 0, stream>>>(W1, W1T, EMBED, FF);
  transpose_f32_bf16<<<(FF * EMBED    + BT - 1) / BT, BT, 0, stream>>>(W2, W2T, FF, EMBED);
  wvsum_t_bf16<<<(HEAD_DIM * EMBED + BT - 1) / BT, BT, 0, stream>>>(Wv, WvT);

  // --- Vsum = x @ Wvsum : (16384 x 1024) @ (1024 x 64) -> flat == A (1024x1024)
  gemm_wmma_bf16<false, 0><<<dim3(HEAD_DIM / 64, ROWS / 32), 32, 0, stream>>>(
      xb, WvT, nullptr, Aact, ROWS, HEAD_DIM, EMBED);

  // --- Y = A @ Wo + bo : (1024 x 1024)
  gemm_wmma_bf16<false, 0><<<dim3(EMBED / 64, AROWS / 32), 32, 0, stream>>>(
      Aact, WoT, bo, Yb, AROWS, EMBED, EMBED);

  // --- H = relu(Y @ W1 + b1) : (1024 x 4096)
  gemm_wmma_bf16<true, 0><<<dim3(FF / 64, AROWS / 32), 32, 0, stream>>>(
      Yb, W1T, b1, Hb, AROWS, FF, EMBED);

  // --- out = broadcast16(H @ W2 + b2) : (1024 x 1024) -> (8, 2048, 1024)
  gemm_wmma_bf16<false, 1><<<dim3(EMBED / 64, AROWS / 32), 32, 0, stream>>>(
      Hb, W2T, b2, out, AROWS, EMBED, FF);
}
